// MiniGPT_30365418783106
// MI455X (gfx1250) — matrix operand order repaired and checked by reading, not compiled, not run on hardware
//
#include <hip/hip_runtime.h>
#include <math.h>

#define Bdim 4
#define Tdim 1024
#define Cdim 768
#define Hdim 12
#define Ldim 6
#define Vdim 32000
#define HDdim 64
#define Mrows (Bdim * Tdim)   // 4096
#define FFdim (4 * Cdim)      // 3072

typedef __attribute__((ext_vector_type(16))) __bf16 v16bf;
typedef __attribute__((ext_vector_type(8)))  float  v8f;
typedef __attribute__((ext_vector_type(4)))  unsigned int u32x4;

union Frag16 { v16bf bf; u32x4 q[2]; };
union Pack8  { u32x4 q; unsigned short s[8]; };

__device__ __forceinline__ unsigned short f2bf(float f) {
  unsigned u = __float_as_uint(f);
  u += 0x7FFFu + ((u >> 16) & 1u);   // round-to-nearest-even
  return (unsigned short)(u >> 16);
}

__device__ __forceinline__ v8f v8f_zero() {
  v8f z;
#pragma unroll
  for (int i = 0; i < 8; ++i) z[i] = 0.f;
  return z;
}

// Async 16-byte global -> LDS copy (gfx1250 GLOBAL_LOAD_ASYNC_TO_LDS_B128,
// tracked by ASYNCcnt). lds_off is the LDS byte offset (low 32 bits of a
// generic pointer to __shared__), gaddr is the 64-bit global address.
__device__ __forceinline__ void async_copy16(unsigned lds_off, const void* gptr) {
  asm volatile("global_load_async_to_lds_b128 %0, %1, off"
               :: "v"(lds_off), "v"((unsigned long long)(size_t)gptr)
               : "memory");
}
__device__ __forceinline__ void async_wait0() {
#if __has_builtin(__builtin_amdgcn_s_wait_asynccnt)
  __builtin_amdgcn_s_wait_asynccnt(0);
#else
  asm volatile("s_wait_asynccnt 0" ::: "memory");
#endif
}

// reductions across the 16 lanes of a C-matrix half (xor masks < 16 stay in-half)
__device__ __forceinline__ float half16_max(float x) {
  x = fmaxf(x, __shfl_xor(x, 8, 32));
  x = fmaxf(x, __shfl_xor(x, 4, 32));
  x = fmaxf(x, __shfl_xor(x, 2, 32));
  x = fmaxf(x, __shfl_xor(x, 1, 32));
  return x;
}
__device__ __forceinline__ float half16_sum(float x) {
  x += __shfl_xor(x, 8, 32);
  x += __shfl_xor(x, 4, 32);
  x += __shfl_xor(x, 2, 32);
  x += __shfl_xor(x, 1, 32);
  return x;
}

// ---------------------------------------------------------------------------
// bf16-WMMA GEMM: C[M,N] = epilogue(A[M,K](bf16) @ B[K,N](f32->bf16))
// block tile 128x128x32, 256 threads (8 wave32), each wave 64x32 of C.
// A tile staged via global_load_async_to_lds_b128; B converted + transposed.
// ---------------------------------------------------------------------------
template <int BIAS, int GELU, int RES, int OBF>
__global__ __launch_bounds__(256) void gemm_bf16_kernel(
    const unsigned short* __restrict__ A, const float* __restrict__ B,
    const float* __restrict__ bias, const float* __restrict__ residual,
    void* __restrict__ Cout, int M, int N, int K)
{
  __shared__ __align__(16) unsigned short As[128][40];  // row-major, 80B rows
  __shared__ __align__(16) unsigned short Bs[128][40];  // N-major (transposed)

  const int tid = threadIdx.x;
  const int bm0 = blockIdx.y * 128;
  const int bn0 = blockIdx.x * 128;
  const int wave = tid >> 5;
  const int lane = tid & 31;
  const int hh   = lane >> 4;
  const int l16  = lane & 15;
  const int wm0  = (wave & 1) * 64;
  const int wn0  = (wave >> 1) * 32;

  v8f acc[4][2];
#pragma unroll
  for (int mt = 0; mt < 4; ++mt)
#pragma unroll
    for (int nt = 0; nt < 2; ++nt) acc[mt][nt] = v8f_zero();

  for (int k0 = 0; k0 < K; k0 += 32) {
    // stage A (bf16) asynchronously: 128 rows x 64B, 16B chunks
    {
      int chunk = tid & 3;        // 0..3
      int r     = tid >> 2;       // 0..63
#pragma unroll
      for (int i = 0; i < 2; ++i) {
        int row = r + i * 64;
        async_copy16((unsigned)(size_t)&As[row][chunk * 8],
                     A + (size_t)(bm0 + row) * K + k0 + chunk * 8);
      }
    }
    // prefetch next B k-slab into GL2
    if (k0 + 32 < K)
      __builtin_prefetch(&B[(size_t)(k0 + 32 + (tid >> 7)) * N + bn0 + (tid & 127)], 0, 0);
    // stage B (f32 -> bf16, transposed to N-major)
    {
      int n  = tid & 127;         // 0..127
      int kk = tid >> 7;          // 0..1
#pragma unroll
      for (int i = 0; i < 16; ++i)
        Bs[n][kk + i * 2] = f2bf(B[(size_t)(k0 + kk + i * 2) * N + bn0 + n]);
    }
    async_wait0();
    __syncthreads();

    Frag16 a[4], b[2];
#pragma unroll
    for (int mt = 0; mt < 4; ++mt) {
      int row = wm0 + mt * 16 + l16;
      a[mt].q[0] = *reinterpret_cast<const u32x4*>(&As[row][8 * hh]);
      a[mt].q[1] = *reinterpret_cast<const u32x4*>(&As[row][16 + 8 * hh]);
    }
#pragma unroll
    for (int nt = 0; nt < 2; ++nt) {
      int col = wn0 + nt * 16 + l16;
      b[nt].q[0] = *reinterpret_cast<const u32x4*>(&Bs[col][8 * hh]);
      b[nt].q[1] = *reinterpret_cast<const u32x4*>(&Bs[col][16 + 8 * hh]);
    }
#pragma unroll
    for (int mt = 0; mt < 4; ++mt)
#pragma unroll
      for (int nt = 0; nt < 2; ++nt)
        acc[mt][nt] = __builtin_amdgcn_wmma_f32_16x16x32_bf16(
            false, a[mt].bf, false, b[nt].bf, (short)0, acc[mt][nt], false, false);
    __syncthreads();
  }

#pragma unroll
  for (int nt = 0; nt < 2; ++nt) {
    int col = bn0 + wn0 + nt * 16 + l16;
    float bval = BIAS ? bias[col] : 0.f;
#pragma unroll
    for (int mt = 0; mt < 4; ++mt)
#pragma unroll
      for (int vv = 0; vv < 8; ++vv) {
        int row = bm0 + wm0 + mt * 16 + vv + 8 * hh;
        float val = acc[mt][nt][vv] + bval;
        if (GELU) val = 0.5f * val * (1.f + erff(val * 0.70710678118f));
        size_t o = (size_t)row * N + col;
        if (RES) val += residual[o];
        if (OBF) ((unsigned short*)Cout)[o] = f2bf(val);
        else     ((float*)Cout)[o] = val;
      }
  }
}

// ---------------------------------------------------------------------------
// Flash attention on bf16 q/k/v: one block (4 wave32) per (b, h, 64-q block).
// Q/K tiles staged with async LDS copies; V transposed through LDS.
// Causal mask applied only on the diagonal key block (scalar-uniform branch).
// ---------------------------------------------------------------------------
__global__ __launch_bounds__(128) void attn_kernel(
    const unsigned short* __restrict__ qp, const unsigned short* __restrict__ kp,
    const unsigned short* __restrict__ vp, unsigned short* __restrict__ y)
{
  __shared__ __align__(16) unsigned short Qs[64][72];      // [q][d]
  __shared__ __align__(16) unsigned short Ks[64][72];      // [key][d]
  __shared__ __align__(16) unsigned short Vt[64][72];      // [d][key]
  __shared__ __align__(16) unsigned short Ps[4][16][72];   // per-wave P tile

  const int tid  = threadIdx.x;
  const int qblk = blockIdx.x;
  const int h    = blockIdx.y;
  const int b    = blockIdx.z;
  const int wave = tid >> 5;
  const int lane = tid & 31;
  const int hh   = lane >> 4;
  const int l16  = lane & 15;
  const int q0   = wave * 16;
  const size_t base = ((size_t)b * Tdim) * Cdim + (size_t)h * HDdim;

  // async-stage Q block: 64 rows x 128B (8 chunks of 16B)
  {
    int chunk = tid & 7;   // 0..7
    int r     = tid >> 3;  // 0..15
#pragma unroll
    for (int i = 0; i < 4; ++i) {
      int row = r + i * 16;
      async_copy16((unsigned)(size_t)&Qs[row][chunk * 8],
                   qp + base + (size_t)(qblk * 64 + row) * Cdim + chunk * 8);
    }
  }

  float m_r[8], l_r[8], alph[8];
  v8f O[4];
#pragma unroll
  for (int vv = 0; vv < 8; ++vv) { m_r[vv] = -INFINITY; l_r[vv] = 0.f; }
#pragma unroll
  for (int t = 0; t < 4; ++t) O[t] = v8f_zero();

  for (int jblk = 0; jblk <= qblk; ++jblk) {
    __syncthreads();   // previous iteration's compute done; safe to overwrite
    // K tile: async 16B copies
    {
      int chunk = tid & 7;
      int r     = tid >> 3;
#pragma unroll
      for (int i = 0; i < 4; ++i) {
        int row = r + i * 16;
        async_copy16((unsigned)(size_t)&Ks[row][chunk * 8],
                     kp + base + (size_t)(jblk * 64 + row) * Cdim + chunk * 8);
      }
    }
    // V tile: vector global loads (8 bf16 per b128), transpose via ds stores
    {
      int chunk = tid & 7;   // d-chunk of 8
      int r     = tid >> 3;  // 0..15
#pragma unroll
      for (int i = 0; i < 4; ++i) {
        int row = r + i * 16;
        Pack8 p;
        p.q = *reinterpret_cast<const u32x4*>(
            vp + base + (size_t)(jblk * 64 + row) * Cdim + chunk * 8);
#pragma unroll
        for (int j = 0; j < 8; ++j) Vt[chunk * 8 + j][row] = p.s[j];
      }
    }
    async_wait0();       // covers this wave's K copies (and Q on first iter)
    __syncthreads();

    // S = Q @ K^T  (HD=64 -> two k-steps of 32)
    Frag16 qa[2];
#pragma unroll
    for (int ks = 0; ks < 2; ++ks) {
      int row = q0 + l16;
      qa[ks].q[0] = *reinterpret_cast<const u32x4*>(&Qs[row][ks * 32 + 8 * hh]);
      qa[ks].q[1] = *reinterpret_cast<const u32x4*>(&Qs[row][ks * 32 + 16 + 8 * hh]);
    }
    v8f S[4];
#pragma unroll
    for (int t = 0; t < 4; ++t) {
      v8f s = v8f_zero();
#pragma unroll
      for (int ks = 0; ks < 2; ++ks) {
        Frag16 kb;
        int col = t * 16 + l16;
        kb.q[0] = *reinterpret_cast<const u32x4*>(&Ks[col][ks * 32 + 8 * hh]);
        kb.q[1] = *reinterpret_cast<const u32x4*>(&Ks[col][ks * 32 + 16 + 8 * hh]);
        s = __builtin_amdgcn_wmma_f32_16x16x32_bf16(
            false, qa[ks].bf, false, kb.bf, (short)0, s, false, false);
      }
      S[t] = s;
    }

    // scale (always); causal mask only on the diagonal block (uniform branch)
#pragma unroll
    for (int t = 0; t < 4; ++t)
#pragma unroll
      for (int vv = 0; vv < 8; ++vv) S[t][vv] *= 0.125f;   // 1/sqrt(64)
    if (jblk == qblk) {
#pragma unroll
      for (int t = 0; t < 4; ++t)
#pragma unroll
        for (int vv = 0; vv < 8; ++vv) {
          int qi = q0 + vv + 8 * hh;
          int ki = t * 16 + l16;
          if (ki > qi) S[t][vv] = -INFINITY;
        }
    }

    // online softmax update
#pragma unroll
    for (int vv = 0; vv < 8; ++vv) {
      float mx = fmaxf(fmaxf(S[0][vv], S[1][vv]), fmaxf(S[2][vv], S[3][vv]));
      mx = half16_max(mx);
      float mn = fmaxf(m_r[vv], mx);
      float p0 = __expf(S[0][vv] - mn), p1 = __expf(S[1][vv] - mn);
      float p2 = __expf(S[2][vv] - mn), p3 = __expf(S[3][vv] - mn);
      S[0][vv] = p0; S[1][vv] = p1; S[2][vv] = p2; S[3][vv] = p3;
      float sm = half16_sum(p0 + p1 + p2 + p3);
      float a = __expf(m_r[vv] - mn);
      l_r[vv] = l_r[vv] * a + sm;
      m_r[vv] = mn;
      alph[vv] = a;
    }
#pragma unroll
    for (int t = 0; t < 4; ++t)
#pragma unroll
      for (int vv = 0; vv < 8; ++vv) O[t][vv] *= alph[vv];

    // P: C-layout regs -> per-wave LDS tile -> A fragments
#pragma unroll
    for (int t = 0; t < 4; ++t)
#pragma unroll
      for (int vv = 0; vv < 8; ++vv)
        Ps[wave][vv + 8 * hh][t * 16 + l16] = f2bf(S[t][vv]);

    Frag16 pa[2];
#pragma unroll
    for (int ks = 0; ks < 2; ++ks) {
      pa[ks].q[0] = *reinterpret_cast<const u32x4*>(&Ps[wave][l16][ks * 32 + 8 * hh]);
      pa[ks].q[1] = *reinterpret_cast<const u32x4*>(&Ps[wave][l16][ks * 32 + 16 + 8 * hh]);
    }
#pragma unroll
    for (int t = 0; t < 4; ++t)
#pragma unroll
      for (int ks = 0; ks < 2; ++ks) {
        Frag16 vb;
        int col = t * 16 + l16;   // d tile
        vb.q[0] = *reinterpret_cast<const u32x4*>(&Vt[col][ks * 32 + 8 * hh]);
        vb.q[1] = *reinterpret_cast<const u32x4*>(&Vt[col][ks * 32 + 16 + 8 * hh]);
        O[t] = __builtin_amdgcn_wmma_f32_16x16x32_bf16(
            false, pa[ks].bf, false, vb.bf, (short)0, O[t], false, false);
      }
  }

#pragma unroll
  for (int t = 0; t < 4; ++t)
#pragma unroll
    for (int vv = 0; vv < 8; ++vv) {
      int row = qblk * 64 + q0 + vv + 8 * hh;
      int d   = t * 16 + l16;
      y[base + (size_t)row * Cdim + d] = f2bf(O[t][vv] / l_r[vv]);
    }
}

// ---------------------------------------------------------------------------
// LayerNorm over C=768, one block per row; emits bf16 for the GEMM A operand
// ---------------------------------------------------------------------------
__global__ __launch_bounds__(256) void ln_kernel(
    const float* __restrict__ x, const float* __restrict__ g,
    const float* __restrict__ bb, unsigned short* __restrict__ out)
{
  __shared__ float red[256];
  int row = blockIdx.x, tid = threadIdx.x;
  const float* xr = x + (size_t)row * Cdim;
  float v0 = xr[tid], v1 = xr[tid + 256], v2 = xr[tid + 512];
  red[tid] = v0 + v1 + v2;
  __syncthreads();
  for (int s = 128; s > 0; s >>= 1) { if (tid < s) red[tid] += red[tid + s]; __syncthreads(); }
  float mu = red[0] * (1.f / Cdim);
  __syncthreads();
  float d0 = v0 - mu, d1 = v1 - mu, d2 = v2 - mu;
  red[tid] = d0 * d0 + d1 * d1 + d2 * d2;
  __syncthreads();
  for (int s = 128; s > 0; s >>= 1) { if (tid < s) red[tid] += red[tid + s]; __syncthreads(); }
  float rstd = rsqrtf(red[0] * (1.f / Cdim) + 1e-5f);
  unsigned short* orow = out + (size_t)row * Cdim;
  orow[tid]       = f2bf(d0 * rstd * g[tid]       + bb[tid]);
  orow[tid + 256] = f2bf(d1 * rstd * g[tid + 256] + bb[tid + 256]);
  orow[tid + 512] = f2bf(d2 * rstd * g[tid + 512] + bb[tid + 512]);
}

__global__ __launch_bounds__(256) void embed_kernel(
    const int* __restrict__ idx, const float* __restrict__ tok,
    const float* __restrict__ pos, float* __restrict__ x)
{
  int row = blockIdx.x, tid = threadIdx.x;
  int t = row & (Tdim - 1);
  int token = idx[row];
  const float* te = tok + (size_t)token * Cdim;
  const float* pe = pos + (size_t)t * Cdim;
  float* xr = x + (size_t)row * Cdim;
  for (int c = tid; c < Cdim; c += 256) xr[c] = te[c] + pe[c];
}

__global__ __launch_bounds__(256) void nll_kernel(
    const float* __restrict__ logits, const int* __restrict__ targets,
    float* __restrict__ nll)
{
  __shared__ float red[256];
  int row = blockIdx.x, tid = threadIdx.x;
  const float* lr = logits + (size_t)row * Vdim;
  float mx = -INFINITY;
  for (int c = tid; c < Vdim; c += 256) mx = fmaxf(mx, lr[c]);
  red[tid] = mx; __syncthreads();
  for (int s = 128; s > 0; s >>= 1) { if (tid < s) red[tid] = fmaxf(red[tid], red[tid + s]); __syncthreads(); }
  mx = red[0]; __syncthreads();
  float sm = 0.f;
  for (int c = tid; c < Vdim; c += 256) sm += __expf(lr[c] - mx);
  red[tid] = sm; __syncthreads();
  for (int s = 128; s > 0; s >>= 1) { if (tid < s) red[tid] += red[tid + s]; __syncthreads(); }
  if (tid == 0) nll[row] = mx + logf(red[0]) - lr[targets[row]];
}

__global__ __launch_bounds__(256) void loss_reduce_kernel(
    const float* __restrict__ nll, float* __restrict__ loss)
{
  __shared__ float red[256];
  int tid = threadIdx.x;
  float s = 0.f;
  for (int i = tid; i < Mrows; i += 256) s += nll[i];
  red[tid] = s; __syncthreads();
  for (int st = 128; st > 0; st >>= 1) { if (tid < st) red[tid] += red[tid + st]; __syncthreads(); }
  if (tid == 0) loss[0] = red[0] / (float)Mrows;
}

extern "C" void kernel_launch(void* const* d_in, const int* in_sizes, int n_in,
                              void* d_out, int out_size, void* d_ws, size_t ws_size,
                              hipStream_t stream)
{
  (void)in_sizes; (void)n_in; (void)ws_size;
  const int*   idx     = (const int*)d_in[0];
  const int*   targets = (const int*)d_in[1];
  const float* tok_emb = (const float*)d_in[2];
  const float* pos_emb = (const float*)d_in[3];
  const float* Wq      = (const float*)d_in[4];
  const float* Wk      = (const float*)d_in[5];
  const float* Wv      = (const float*)d_in[6];
  const float* Wo      = (const float*)d_in[7];
  const float* bo      = (const float*)d_in[8];
  const float* ln1_g   = (const float*)d_in[9];
  const float* ln1_b   = (const float*)d_in[10];
  const float* ln2_g   = (const float*)d_in[11];
  const float* ln2_b   = (const float*)d_in[12];
  const float* W1      = (const float*)d_in[13];
  const float* b1      = (const float*)d_in[14];
  const float* W2      = (const float*)d_in[15];
  const float* b2      = (const float*)d_in[16];
  const float* lnf_g   = (const float*)d_in[17];
  const float* lnf_b   = (const float*)d_in[18];
  const float* Whead   = (const float*)d_in[19];

  float* logits = (float*)d_out;
  float* loss   = logits + ((size_t)out_size - 1);

  // workspace partition
  char* w = (char*)d_ws;
  const size_t MCf = (size_t)Mrows * Cdim * sizeof(float);
  const size_t MCh = (size_t)Mrows * Cdim * sizeof(unsigned short);
  const size_t MFh = (size_t)Mrows * FFdim * sizeof(unsigned short);
  float*          x     = (float*)w;                 w += MCf;
  unsigned short* lnbuf = (unsigned short*)w;        w += MCh;
  unsigned short* qb    = (unsigned short*)w;        w += MCh;
  unsigned short* kb    = (unsigned short*)w;        w += MCh;
  unsigned short* vb    = (unsigned short*)w;        w += MCh;
  unsigned short* yb    = (unsigned short*)w;        w += MCh;
  unsigned short* hid   = (unsigned short*)w;        w += MFh;
  float*          nll   = (float*)w;

  dim3 gCC(Cdim / 128, Mrows / 128);     // N=768
  dim3 gFF(FFdim / 128, Mrows / 128);    // N=3072
  dim3 gHD(Vdim / 128, Mrows / 128);     // N=32000

  embed_kernel<<<Mrows, 256, 0, stream>>>(idx, tok_emb, pos_emb, x);

  for (int i = 0; i < Ldim; ++i) {
    const float* wq = Wq + (size_t)i * Cdim * Cdim;
    const float* wk = Wk + (size_t)i * Cdim * Cdim;
    const float* wv = Wv + (size_t)i * Cdim * Cdim;
    const float* wo = Wo + (size_t)i * Cdim * Cdim;
    const float* w1 = W1 + (size_t)i * Cdim * FFdim;
    const float* w2 = W2 + (size_t)i * FFdim * Cdim;

    ln_kernel<<<Mrows, 256, 0, stream>>>(x, ln1_g + i * Cdim, ln1_b + i * Cdim, lnbuf);
    gemm_bf16_kernel<0,0,0,1><<<gCC, 256, 0, stream>>>(lnbuf, wq, nullptr, nullptr, qb, Mrows, Cdim, Cdim);
    gemm_bf16_kernel<0,0,0,1><<<gCC, 256, 0, stream>>>(lnbuf, wk, nullptr, nullptr, kb, Mrows, Cdim, Cdim);
    gemm_bf16_kernel<0,0,0,1><<<gCC, 256, 0, stream>>>(lnbuf, wv, nullptr, nullptr, vb, Mrows, Cdim, Cdim);
    attn_kernel<<<dim3(Tdim / 64, Hdim, Bdim), 128, 0, stream>>>(qb, kb, vb, yb);
    gemm_bf16_kernel<1,0,1,0><<<gCC, 256, 0, stream>>>(yb, wo, bo + i * Cdim, x, x, Mrows, Cdim, Cdim);

    ln_kernel<<<Mrows, 256, 0, stream>>>(x, ln2_g + i * Cdim, ln2_b + i * Cdim, lnbuf);
    gemm_bf16_kernel<1,1,0,1><<<gFF, 256, 0, stream>>>(lnbuf, w1, b1 + i * FFdim, nullptr, hid, Mrows, FFdim, Cdim);
    gemm_bf16_kernel<1,0,1,0><<<gCC, 256, 0, stream>>>(hid, w2, b2 + i * Cdim, x, x, Mrows, Cdim, FFdim);
  }

  ln_kernel<<<Mrows, 256, 0, stream>>>(x, lnf_g, lnf_b, lnbuf);
  gemm_bf16_kernel<0,0,0,0><<<gHD, 256, 0, stream>>>(lnbuf, Whead, nullptr, nullptr, logits, Mrows, Vdim, Cdim);
  nll_kernel<<<Mrows, 256, 0, stream>>>(logits, targets, nll);
  loss_reduce_kernel<<<1, 256, 0, stream>>>(nll, loss);
}
